// Attention_72593537237211
// MI455X (gfx1250) — matrix operand order repaired and checked by reading, not compiled, not run on hardware
//
#include <hip/hip_runtime.h>
#include <hip/hip_bf16.h>

typedef __attribute__((ext_vector_type(16))) __bf16 v16bf;
typedef __attribute__((ext_vector_type(8)))  float  v8f;

#define B_DIM 64
#define S_DIM 2048
#define E_DIM 512
#define D_DIM 512
#define NT 32   /* D/16 n-tiles */
#define KT 16   /* E/32 k-steps */

__device__ __forceinline__ float fast_tanh(float x)
{
#if __has_builtin(__builtin_amdgcn_tanhf)
    return __builtin_amdgcn_tanhf(x);   // v_tanh_f32 (gfx1250 TRANS op)
#else
    return tanhf(x);
#endif
}

// ---------------------------------------------------------------------------
// pre[b][k] = sum_d hidden[b][d] * Wh[d][k] + attn_b[k]   (Wh = attn_W[0:512])
// ---------------------------------------------------------------------------
__global__ __launch_bounds__(256)
void k_hproj(const float* __restrict__ hidden, const float* __restrict__ W,
             const float* __restrict__ bias, float* __restrict__ pre)
{
    int b  = blockIdx.x;
    int k0 = threadIdx.x * 2;
    float acc0 = 0.f, acc1 = 0.f;
    for (int d = 0; d < D_DIM; ++d) {
        float h = hidden[b * D_DIM + d];
        acc0 += h * W[d * D_DIM + k0];
        acc1 += h * W[d * D_DIM + k0 + 1];
    }
    pre[b * D_DIM + k0]     = acc0 + bias[k0];
    pre[b * D_DIM + k0 + 1] = acc1 + bias[k0 + 1];
}

// ---------------------------------------------------------------------------
// Pack We (= attn_W[512:1024][:]) to bf16 panels: wsB[nt][k>>5][lane][s], nt=n>>4, j=n&15
// WMMA B fragment: lane L holds column N=L&15, its 16 K values (table order) contiguous.
// ---------------------------------------------------------------------------
__global__ __launch_bounds__(512)
void k_packB(const float* __restrict__ attn_W, __hip_bfloat16* __restrict__ wsB)
{
    int idx = blockIdx.x * blockDim.x + threadIdx.x;   // 0 .. 512*512-1
    int n = idx & (D_DIM - 1);
    int k = idx >> 9;
    int nt = n >> 4, j = n & 15;
    wsB[(size_t)nt * (E_DIM * 16) + (k >> 5) * 512 + ((((k >> 3) & 1) * 16 + j) * 16) + ((k & 7) | (((k >> 4) & 1) << 3))] =
        __float2bfloat16(attn_W[(size_t)(E_DIM + k) * D_DIM + n]);
}

// ---------------------------------------------------------------------------
// Load one 16x32 A fragment (ISA 16-bit A layout) from a f32 row-major tile:
// lane (&15) selects M row; lanes>=16 take K-chunks {kb+8..15, kb+24..31}.
// ---------------------------------------------------------------------------
__device__ __forceinline__ v16bf load_afrag(const float* __restrict__ rowp,
                                            int ks, int hi)
{
    const float* p0 = rowp + ks * 32 + hi * 8;
    const float* p1 = p0 + 16;
    float4 f0 = ((const float4*)p0)[0];
    float4 f1 = ((const float4*)p0)[1];
    float4 g0 = ((const float4*)p1)[0];
    float4 g1 = ((const float4*)p1)[1];
    v16bf a;
    a[0]=(__bf16)f0.x;  a[1]=(__bf16)f0.y;  a[2]=(__bf16)f0.z;  a[3]=(__bf16)f0.w;
    a[4]=(__bf16)f1.x;  a[5]=(__bf16)f1.y;  a[6]=(__bf16)f1.z;  a[7]=(__bf16)f1.w;
    a[8]=(__bf16)g0.x;  a[9]=(__bf16)g0.y;  a[10]=(__bf16)g0.z; a[11]=(__bf16)g0.w;
    a[12]=(__bf16)g1.x; a[13]=(__bf16)g1.y; a[14]=(__bf16)g1.z; a[15]=(__bf16)g1.w;
    return a;
}

// ---------------------------------------------------------------------------
// Main fused kernel. Rows m = s*64+b of enc viewed as (S*B) x E.
// Each wave owns TWO 16-row M-tiles: B fragments are loaded once and feed two
// WMMAs (halves B-operand bandwidth, doubles accumulator-chain ILP).
// scores[m] = v_w . tanh(pre[b] + enc_row @ We)
// ---------------------------------------------------------------------------
__global__ __launch_bounds__(256)
void k_attn_main(const float* __restrict__ enc,
                 const float* __restrict__ pre,
                 const __hip_bfloat16* __restrict__ wsB,
                 const float* __restrict__ v_w,
                 float* __restrict__ scores)
{
    const int lane = threadIdx.x & 31;
    const int wave = threadIdx.x >> 5;
    const int hi   = lane >> 4;      // half-wave select
    const int ncol = lane & 15;

    const int tile0  = (blockIdx.x * 8 + wave) * 2;    // 0..8190 step 2
    const int mBase0 = tile0 * 16;
    const int mBase1 = mBase0 + 16;

    // ---- A tiles: 2 x (16 rows x 512 K) f32 -> bf16 register fragments ----
    const float* rowp0 = enc + (size_t)(mBase0 + ncol) * E_DIM;
    const float* rowp1 = enc + (size_t)(mBase1 + ncol) * E_DIM;
    v16bf afrag0[KT], afrag1[KT];
#pragma unroll
    for (int ks = 0; ks < KT; ++ks) afrag0[ks] = load_afrag(rowp0, ks, hi);
#pragma unroll
    for (int ks = 0; ks < KT; ++ks) afrag1[ks] = load_afrag(rowp1, ks, hi);

    float spart0[8], spart1[8];
#pragma unroll
    for (int i = 0; i < 8; ++i) { spart0[i] = 0.f; spart1[i] = 0.f; }

    for (int n = 0; n < NT; ++n) {
        // C init = pre (h_proj + bias) so WMMA yields h_proj + e_proj + b.
        // C layout: lanes 0-15 VGPR i -> (M=i, N=lane); lanes 16-31 -> (M=8+i).
        v8f c0, c1;
#pragma unroll
        for (int i = 0; i < 8; ++i) {
            int b0 = (mBase0 + i + hi * 8) & (B_DIM - 1);
            int b1 = (mBase1 + i + hi * 8) & (B_DIM - 1);
            c0[i] = pre[b0 * D_DIM + n * 16 + ncol];
            c1[i] = pre[b1 * D_DIM + n * 16 + ncol];
        }

        const v16bf* bp = (const v16bf*)(wsB + (size_t)n * (E_DIM * 16));
#pragma unroll
        for (int ks = 0; ks < KT; ++ks) {
            v16bf bfrag = bp[ks * 32 + lane];           // lane's B frag (col N=lane&15) for k-step ks
            c0 = __builtin_amdgcn_wmma_f32_16x16x32_bf16(
                     false, afrag0[ks], false, bfrag, (short)0, c0, false, false);
            c1 = __builtin_amdgcn_wmma_f32_16x16x32_bf16(
                     false, afrag1[ks], false, bfrag, (short)0, c1, false, false);
        }

        float vw = v_w[n * 16 + ncol];
#pragma unroll
        for (int i = 0; i < 8; ++i) {
            spart0[i] += fast_tanh(c0[i]) * vw;
            spart1[i] += fast_tanh(c1[i]) * vw;
        }
    }

    // Reduce over the 16 N-lanes of each half-wave (rows live per half).
#pragma unroll
    for (int off = 1; off < 16; off <<= 1) {
#pragma unroll
        for (int i = 0; i < 8; ++i) {
            spart0[i] += __shfl_xor(spart0[i], off, 32);
            spart1[i] += __shfl_xor(spart1[i], off, 32);
        }
    }
    if (ncol == 0) {
#pragma unroll
        for (int i = 0; i < 8; ++i) {
            scores[mBase0 + hi * 8 + i] = spart0[i];
            scores[mBase1 + hi * 8 + i] = spart1[i];
        }
    }
}

// ---------------------------------------------------------------------------
// Softmax over S for each b: out[b][s] = softmax_s(scores[s*64+b])
// ---------------------------------------------------------------------------
__global__ __launch_bounds__(256)
void k_softmax(const float* __restrict__ sc, float* __restrict__ out)
{
    __shared__ float red[256];
    int b = blockIdx.x, t = threadIdx.x;
    float vals[8];
    float mx = -3.4e38f;
#pragma unroll
    for (int j = 0; j < 8; ++j) {
        int s = t + j * 256;
        vals[j] = sc[s * B_DIM + b];
        mx = fmaxf(mx, vals[j]);
    }
    red[t] = mx; __syncthreads();
    for (int o = 128; o > 0; o >>= 1) {
        if (t < o) red[t] = fmaxf(red[t], red[t + o]);
        __syncthreads();
    }
    mx = red[0]; __syncthreads();
    float sum = 0.f;
#pragma unroll
    for (int j = 0; j < 8; ++j) { vals[j] = __expf(vals[j] - mx); sum += vals[j]; }
    red[t] = sum; __syncthreads();
    for (int o = 128; o > 0; o >>= 1) {
        if (t < o) red[t] += red[t + o];
        __syncthreads();
    }
    float inv = 1.f / red[0];
#pragma unroll
    for (int j = 0; j < 8; ++j)
        out[b * S_DIM + t + j * 256] = vals[j] * inv;
}

// ---------------------------------------------------------------------------
extern "C" void kernel_launch(void* const* d_in, const int* in_sizes, int n_in,
                              void* d_out, int out_size, void* d_ws, size_t ws_size,
                              hipStream_t stream)
{
    const float* hidden = (const float*)d_in[0];
    const float* enc    = (const float*)d_in[1];
    const float* attn_W = (const float*)d_in[2];
    const float* attn_b = (const float*)d_in[3];
    const float* v_w    = (const float*)d_in[4];
    float* out = (float*)d_out;

    char* ws = (char*)d_ws;
    float*          pre    = (float*)ws;                         // 64*512*4   = 128 KB
    __hip_bfloat16* wsB    = (__hip_bfloat16*)(ws + (131072));   // 512*512*2  = 512 KB
    float*          scores = (float*)(ws + 131072 + 524288);     // 131072*4   = 512 KB

    k_hproj    <<<B_DIM, 256, 0, stream>>>(hidden, attn_W, attn_b, pre);
    k_packB    <<<512,   512, 0, stream>>>(attn_W, wsB);
    k_attn_main<<<512,   256, 0, stream>>>(enc, pre, wsB, v_w, scores);
    k_softmax  <<<B_DIM, 256, 0, stream>>>(scores, out);
}
